// WindowAttention_13675175870701
// MI455X (gfx1250) — matrix operand order repaired and checked by reading, not compiled, not run on hardware
//
#include <hip/hip_runtime.h>
#include <hip/hip_bf16.h>

typedef __attribute__((ext_vector_type(16))) __bf16 v16bf;
typedef __attribute__((ext_vector_type(2)))  __bf16 v2bf;
typedef __attribute__((ext_vector_type(8)))  float  v8f;
typedef __attribute__((ext_vector_type(4)))  int    v4i;
typedef unsigned short ushort_t;
typedef __attribute__((address_space(1))) v4i as1_v4i;
typedef __attribute__((address_space(3))) v4i as3_v4i;
typedef __attribute__((address_space(3))) void as3_void;

#define WIN 7
#define NTOK 49           // WIN*WIN
#define NHEAD 6
#define HDIM 32
#define CCH 192           // NHEAD*HDIM
#define NWIN 2048
#define MTOK (NWIN * NTOK)   // 100352, divisible by 64
#define QKSCALE 0.17677669529663687f  // 1/sqrt(32)
#define VSTRIDE (HDIM * 64)           // padded v tile per (win,head): [32][64]

// ---- fp32 -> bf16 conversions ---------------------------------------------
static __device__ inline unsigned bfrnd(float f) {
    unsigned u = __builtin_bit_cast(unsigned, f);
    return u + 0x7FFFu + ((u >> 16) & 1u);   // RNE in bits [31:16]
}

#if __has_builtin(__builtin_amdgcn_cvt_pk_bf16_f32)
static __device__ inline unsigned pk2bf(float a, float b) {
    v2bf r = __builtin_amdgcn_cvt_pk_bf16_f32(a, b);
    return __builtin_bit_cast(unsigned, r);
}
#else
// pack {hi16(rnd(b)), hi16(rnd(a))} with one v_perm_b32
static __device__ inline unsigned pk2bf(float a, float b) {
    return __builtin_amdgcn_perm(bfrnd(a), bfrnd(b), 0x03020706u);
}
#endif

static __device__ inline ushort_t f2bf(float f) {
    return (ushort_t)(bfrnd(f) >> 16);
}

// ---- async global->LDS copy (16B), tracked by ASYNCcnt -------------------
static __device__ inline void async_b128(const ushort_t* g, ushort_t* l) {
#if __has_builtin(__builtin_amdgcn_global_load_async_to_lds_b128)
    __builtin_amdgcn_global_load_async_to_lds_b128(
        (as1_v4i*)g, (as3_v4i*)l, 0, 0);
#else
    unsigned lofs = (unsigned)(__UINTPTR_TYPE__)(as3_void*)l;
    unsigned long long ga = (unsigned long long)(__UINTPTR_TYPE__)g;
    asm volatile("global_load_async_to_lds_b128 %0, %1, off"
                 :: "v"(lofs), "v"(ga) : "memory");
#endif
}

static __device__ inline void wait_async0() {
#if __has_builtin(__builtin_amdgcn_s_wait_asynccnt)
    __builtin_amdgcn_s_wait_asynccnt(0);
#else
    asm volatile("s_wait_asynccnt 0x0" ::: "memory");
#endif
}

// Load a 16-element bf16 WMMA operand from two aligned 16B LDS chunks.
static __device__ inline v16bf ld_op2(const ushort_t* p0, const ushort_t* p1) {
    union { uint4 u[2]; v16bf v; } t;
    t.u[0] = *reinterpret_cast<const uint4*>(p0);
    t.u[1] = *reinterpret_cast<const uint4*>(p1);
    return t.v;
}

// A operand: 16x32 bf16 tile, LDS [row][32]
static __device__ inline v16bf ld_a32(const ushort_t* lds, int tile16, int lane) {
    int half = lane >> 4, m = lane & 15;
    const ushort_t* base = lds + (tile16 * 16 + m) * 32;
    return ld_op2(base + half * 8, base + 16 + half * 8);
}

// B operand: 32x16 bf16 tile, LDS [col][32] (K contiguous per column)
static __device__ inline v16bf ld_b32(const ushort_t* lds, int tile16, int lane) {
    int half = lane >> 4, n = lane & 15;
    const ushort_t* base = lds + (tile16 * 16 + n) * 32;
    return ld_op2(base + half * 8, base + 16 + half * 8);
}

static __device__ inline v8f wmma_bf16(v16bf a, v16bf b, v8f c) {
    return __builtin_amdgcn_wmma_f32_16x16x32_bf16(false, a, false, b, (short)0, c,
                                                   false, false);
}

// Pack 4 fp32 -> 4 bf16 (as uint2)
static __device__ inline uint2 pack4bf(float4 f) {
    return make_uint2(pk2bf(f.x, f.y), pk2bf(f.z, f.w));
}

// ---------------------------------------------------------------------------
// Kernel 0: fp32 weights -> bf16 TRANSPOSED copies  wT[n*192 + k] = w[k*N + n]
// ---------------------------------------------------------------------------
__global__ void wa_cvt_weights(const float* __restrict__ qkv_w,
                               const float* __restrict__ proj_w,
                               ushort_t* __restrict__ qkv_wT,
                               ushort_t* __restrict__ proj_wT) {
    int i = blockIdx.x * blockDim.x + threadIdx.x;
    const int NQ = (3 * CCH) * CCH;   // 110592 : qkv_wT is [576][192]
    const int NP = CCH * CCH;         // 36864  : proj_wT is [192][192]
    if (i < NQ) {
        int n = i / CCH, k = i - n * CCH;
        qkv_wT[i] = f2bf(qkv_w[(size_t)k * (3 * CCH) + n]);
    } else if (i < NQ + NP) {
        int j = i - NQ;
        int n = j / CCH, k = j - n * CCH;
        proj_wT[j] = f2bf(proj_w[(size_t)k * CCH + n]);
    }
}

// ---------------------------------------------------------------------------
// Kernel 1: QKV GEMM, double-buffered: async B tiles + VALU-converted A tiles
// prefetched for step s+1 while WMMAs run on step s.
// ---------------------------------------------------------------------------
__global__ void wa_qkv_gemm(const float* __restrict__ x,
                            const ushort_t* __restrict__ wT,
                            const float* __restrict__ bias,
                            ushort_t* __restrict__ qb,
                            ushort_t* __restrict__ kb,
                            ushort_t* __restrict__ vb) {
    __shared__ ushort_t As[2][64 * 32];
    __shared__ ushort_t Bs[2][64 * 32];   // stored [n][k]

    const int tid = threadIdx.x;
    const int lane = tid & 31;
    const int wave = tid >> 5;
    const int wm = wave & 1, wn = wave >> 1;
    const int rowBase = blockIdx.x * 64;
    const int nBase = blockIdx.y * 64;

    v8f acc[2][2] = {};

    // stage(buf, k0): B via async, A via fp32->bf16 convert.
    // Load phase uses 4 independent temps so the global loads clause
    // behind a single s_wait_loadcnt.
    auto stage = [&](int buf, int k0) {
        #pragma unroll
        for (int i = 0; i < 2; ++i) {
            int e = i * 128 + tid;               // 0..255
            int n = e >> 2, c4 = e & 3;
            async_b128(wT + (size_t)(nBase + n) * CCH + k0 + c4 * 8,
                       &Bs[buf][n * 32 + c4 * 8]);
        }
        float4 fbuf[4];
        #pragma unroll
        for (int i = 0; i < 4; ++i) {
            int e4 = i * 128 + tid;              // 0..511
            int r = e4 >> 3, c4 = e4 & 7;
            fbuf[i] = *reinterpret_cast<const float4*>(
                x + (size_t)(rowBase + r) * CCH + k0 + c4 * 4);
        }
        #pragma unroll
        for (int i = 0; i < 4; ++i) {
            int e4 = i * 128 + tid;
            int r = e4 >> 3, c4 = e4 & 7;
            *reinterpret_cast<uint2*>(&As[buf][r * 32 + c4 * 4]) = pack4bf(fbuf[i]);
        }
    };

    stage(0, 0);
    wait_async0();
    __syncthreads();

    #pragma unroll
    for (int s = 0; s < 6; ++s) {
        int cur = s & 1;
        if (s < 5) stage(cur ^ 1, (s + 1) * 32);

        v16bf a0 = ld_a32(As[cur], 2 * wm + 0, lane);
        v16bf a1 = ld_a32(As[cur], 2 * wm + 1, lane);
        v16bf b0 = ld_b32(Bs[cur], 2 * wn + 0, lane);
        v16bf b1 = ld_b32(Bs[cur], 2 * wn + 1, lane);
        acc[0][0] = wmma_bf16(a0, b0, acc[0][0]);
        acc[0][1] = wmma_bf16(a0, b1, acc[0][1]);
        acc[1][0] = wmma_bf16(a1, b0, acc[1][0]);
        acc[1][1] = wmma_bf16(a1, b1, acc[1][1]);

        if (s < 5) {
            wait_async0();
            __syncthreads();
        }
    }

    const int half = lane >> 4;
    #pragma unroll
    for (int i = 0; i < 2; ++i)
        #pragma unroll
        for (int j = 0; j < 2; ++j) {
            int gcol = nBase + (2 * wn + j) * 16 + (lane & 15);
            int s = gcol / CCH;
            int hh = (gcol % CCH) >> 5;
            int d = gcol & 31;
            float bv = bias[gcol];
            float scl = (s == 0) ? QKSCALE : 1.0f;
            ushort_t* dst = (s == 0) ? qb : ((s == 1) ? kb : vb);
            #pragma unroll
            for (int r = 0; r < 8; ++r) {
                int rl = (2 * wm + i) * 16 + half * 8 + r;
                int tok = rowBase + rl;
                int b = tok / NTOK, n = tok - b * NTOK;
                float val = (acc[i][j][r] + bv) * scl;
                int bh = b * NHEAD + hh;
                size_t oqk = (size_t)bh * (NTOK * HDIM) + n * HDIM + d;
                size_t ov  = (size_t)bh * VSTRIDE + d * 64 + n;
                dst[(s == 2) ? ov : oqk] = f2bf(val);
            }
        }
}

// ---------------------------------------------------------------------------
// Kernel 2: attention core. One wave per (window, head). Async staging.
// ---------------------------------------------------------------------------
__global__ __launch_bounds__(32)
void wa_attn(const ushort_t* __restrict__ qb,
             const ushort_t* __restrict__ kb,
             const ushort_t* __restrict__ vb,
             const float* __restrict__ rpb,
             ushort_t* __restrict__ ao) {
    __shared__ ushort_t sq[64 * 32];
    __shared__ ushort_t sk[64 * 32];
    __shared__ ushort_t sv[32 * 64];   // [d][key]
    __shared__ float    ss[64 * 64];
    __shared__ ushort_t sp[64 * 64];

    const int lane = threadIdx.x;
    const int b = blockIdx.x / NHEAD;
    const int h = blockIdx.x % NHEAD;
    const int bh = b * NHEAD + h;
    const size_t qkBase = (size_t)bh * (NTOK * HDIM);
    const size_t vBase  = (size_t)bh * VSTRIDE;

    // q/k staging: 49*32 = 1568 bf16 = 196 x b128 async copies
    #pragma unroll
    for (int i = 0; i < 7; ++i) {
        int idx4 = i * 32 + lane;                // 0..223
        if (idx4 < 196) {
            async_b128(qb + qkBase + idx4 * 8, &sq[idx4 * 8]);
            async_b128(kb + qkBase + idx4 * 8, &sk[idx4 * 8]);
        }
    }
    // zero pads (rows 49..63)
    const uint4 zero4 = make_uint4(0u, 0u, 0u, 0u);
    #pragma unroll
    for (int i = 0; i < 2; ++i) {
        int idx4 = 196 + i * 32 + lane;
        if (idx4 < 256) {
            *reinterpret_cast<uint4*>(&sq[idx4 * 8]) = zero4;
            *reinterpret_cast<uint4*>(&sk[idx4 * 8]) = zero4;
        }
    }
    // v staging from padded [d][64] layout: async rows + scalar tail + zero pads
    {
        int d = lane;                            // 32 lanes == 32 d-rows
        const ushort_t* src = vb + vBase + d * 64;
        ushort_t* dstr = &sv[d * 64];
        #pragma unroll
        for (int c4 = 0; c4 < 6; ++c4)           // n = 0..47
            async_b128(src + c4 * 8, dstr + c4 * 8);
        dstr[48] = src[48];                      // n = 48
        #pragma unroll
        for (int n = 49; n < 64; ++n) dstr[n] = 0;
    }
    wait_async0();
    __syncthreads();

    const int half = lane >> 4;
    // ---- S = q k^T, add relative position bias ----
    for (int tm = 0; tm < 4; ++tm) {
        v16bf aop = ld_a32(sq, tm, lane);
        for (int tn = 0; tn < 4; ++tn) {
            v16bf bop = ld_b32(sk, tn, lane);
            v8f c = {};
            c = wmma_bf16(aop, bop, c);
            int col = tn * 16 + (lane & 15);
            int qi = col / WIN, qj = col - qi * WIN;
            #pragma unroll
            for (int r = 0; r < 8; ++r) {
                int row = tm * 16 + half * 8 + r;
                float val = c[r];
                if (row < NTOK && col < NTOK) {
                    int pi = row / WIN, pj = row - pi * WIN;
                    int idx = (pi - qi + WIN - 1) * (2 * WIN - 1) + (pj - qj + WIN - 1);
                    val += rpb[idx * NHEAD + h];
                }
                ss[row * 64 + col] = val;
            }
        }
    }
    __syncthreads();

    // ---- softmax over valid 49 columns; rows/cols >=49 zeroed in P ----
    for (int rr = 0; rr < 2; ++rr) {
        int row = lane + rr * 32;
        if (row < NTOK) {
            float mx = -1e30f;
            for (int c2 = 0; c2 < NTOK; ++c2) mx = fmaxf(mx, ss[row * 64 + c2]);
            float sum = 0.f;
            for (int c2 = 0; c2 < NTOK; ++c2) {
                float e = __expf(ss[row * 64 + c2] - mx);
                ss[row * 64 + c2] = e;
                sum += e;
            }
            float inv = 1.0f / sum;
            for (int c2 = 0; c2 < 64; c2 += 2) {
                float a = (c2 < NTOK)      ? ss[row * 64 + c2] * inv     : 0.f;
                float bb = (c2 + 1 < NTOK) ? ss[row * 64 + c2 + 1] * inv : 0.f;
                *reinterpret_cast<unsigned*>(&sp[row * 64 + c2]) = pk2bf(a, bb);
            }
        } else {
            for (int c2 = 0; c2 < 64; c2 += 2)
                *reinterpret_cast<unsigned*>(&sp[row * 64 + c2]) = 0u;
        }
    }
    __syncthreads();

    // ---- out = P v ----
    for (int tm = 0; tm < 4; ++tm) {
        for (int tn = 0; tn < 2; ++tn) {
            v8f c = {};
            #pragma unroll
            for (int kk = 0; kk < 2; ++kk) {
                const ushort_t* pa = sp + (tm * 16 + (lane & 15)) * 64 + kk * 32;
                v16bf aop = ld_op2(pa + half * 8, pa + 16 + half * 8);
                const ushort_t* pb = sv + (tn * 16 + (lane & 15)) * 64 + kk * 32;
                v16bf bop = ld_op2(pb + half * 8, pb + 16 + half * 8);
                c = wmma_bf16(aop, bop, c);
            }
            int dcol = tn * 16 + (lane & 15);
            #pragma unroll
            for (int r = 0; r < 8; ++r) {
                int row = tm * 16 + half * 8 + r;
                if (row < NTOK)
                    ao[(size_t)(b * NTOK + row) * CCH + h * HDIM + dcol] = f2bf(c[r]);
            }
        }
    }
}

// ---------------------------------------------------------------------------
// Kernel 3: output projection GEMM, double-buffered, A and B both async.
// ---------------------------------------------------------------------------
__global__ void wa_proj_gemm(const ushort_t* __restrict__ ain,
                             const ushort_t* __restrict__ wT,
                             const float* __restrict__ bias,
                             float* __restrict__ out) {
    __shared__ ushort_t As[2][64 * 32];
    __shared__ ushort_t Bs[2][64 * 32];

    const int tid = threadIdx.x;
    const int lane = tid & 31;
    const int wave = tid >> 5;
    const int wm = wave & 1, wn = wave >> 1;
    const int rowBase = blockIdx.x * 64;
    const int nBase = blockIdx.y * 64;

    v8f acc[2][2] = {};

    auto stage = [&](int buf, int k0) {
        #pragma unroll
        for (int i = 0; i < 2; ++i) {
            int e = i * 128 + tid;
            int r = e >> 2, c4 = e & 3;
            async_b128(ain + (size_t)(rowBase + r) * CCH + k0 + c4 * 8,
                       &As[buf][r * 32 + c4 * 8]);
        }
        #pragma unroll
        for (int i = 0; i < 2; ++i) {
            int e = i * 128 + tid;
            int n = e >> 2, c4 = e & 3;
            async_b128(wT + (size_t)(nBase + n) * CCH + k0 + c4 * 8,
                       &Bs[buf][n * 32 + c4 * 8]);
        }
    };

    stage(0, 0);
    wait_async0();
    __syncthreads();

    #pragma unroll
    for (int s = 0; s < 6; ++s) {
        int cur = s & 1;
        if (s < 5) stage(cur ^ 1, (s + 1) * 32);

        v16bf a0 = ld_a32(As[cur], 2 * wm + 0, lane);
        v16bf a1 = ld_a32(As[cur], 2 * wm + 1, lane);
        v16bf b0 = ld_b32(Bs[cur], 2 * wn + 0, lane);
        v16bf b1 = ld_b32(Bs[cur], 2 * wn + 1, lane);
        acc[0][0] = wmma_bf16(a0, b0, acc[0][0]);
        acc[0][1] = wmma_bf16(a0, b1, acc[0][1]);
        acc[1][0] = wmma_bf16(a1, b0, acc[1][0]);
        acc[1][1] = wmma_bf16(a1, b1, acc[1][1]);

        if (s < 5) {
            wait_async0();
            __syncthreads();
        }
    }

    const int half = lane >> 4;
    #pragma unroll
    for (int i = 0; i < 2; ++i)
        #pragma unroll
        for (int j = 0; j < 2; ++j) {
            int gcol = nBase + (2 * wn + j) * 16 + (lane & 15);
            float bv = bias[gcol];
            #pragma unroll
            for (int r = 0; r < 8; ++r) {
                int rl = (2 * wm + i) * 16 + half * 8 + r;
                out[(size_t)(rowBase + rl) * CCH + gcol] = acc[i][j][r] + bv;
            }
        }
}

// ---------------------------------------------------------------------------
extern "C" void kernel_launch(void* const* d_in, const int* in_sizes, int n_in,
                              void* d_out, int out_size, void* d_ws, size_t ws_size,
                              hipStream_t stream) {
    const float* x      = (const float*)d_in[0];
    // d_in[1] = q_global: unused by the reference computation
    const float* qkv_w  = (const float*)d_in[2];
    const float* qkv_b  = (const float*)d_in[3];
    const float* proj_w = (const float*)d_in[4];
    const float* proj_b = (const float*)d_in[5];
    const float* rpb    = (const float*)d_in[6];
    float* out = (float*)d_out;

    // Workspace layout (ushort elements)
    const size_t QKBUF = (size_t)NWIN * NHEAD * NTOK * HDIM;  // 19267584
    const size_t VBUF  = (size_t)NWIN * NHEAD * VSTRIDE;      // 25165824
    ushort_t* ws = (ushort_t*)d_ws;
    ushort_t* qb        = ws;
    ushort_t* kb        = qb + QKBUF;
    ushort_t* vb        = kb + QKBUF;
    ushort_t* attn_out  = vb + VBUF;                          // MTOK*CCH
    ushort_t* qkv_wT    = attn_out + (size_t)MTOK * CCH;
    ushort_t* proj_wT   = qkv_wT + (size_t)(3 * CCH) * CCH;

    // 0) weights -> bf16 (transposed [N][K])
    {
        int tot = (3 * CCH) * CCH + CCH * CCH;
        wa_cvt_weights<<<(tot + 255) / 256, 256, 0, stream>>>(qkv_w, proj_w,
                                                              qkv_wT, proj_wT);
    }
    // 1) QKV GEMM + scatter
    {
        dim3 grid(MTOK / 64, (3 * CCH) / 64);   // 1568 x 9
        wa_qkv_gemm<<<grid, 128, 0, stream>>>(x, qkv_wT, qkv_b, qb, kb, vb);
    }
    // 2) attention core: one wave per (window, head)
    {
        wa_attn<<<NWIN * NHEAD, 32, 0, stream>>>(qb, kb, vb, rpb, attn_out);
    }
    // 3) output projection
    {
        dim3 grid(MTOK / 64, CCH / 64);         // 1568 x 3
        wa_proj_gemm<<<grid, 128, 0, stream>>>(attn_out, proj_wT, proj_b, out);
    }
}
